// MultiAttention_75737453298397
// MI455X (gfx1250) — matrix operand order repaired and checked by reading, not compiled, not run on hardware
//
#include <hip/hip_runtime.h>
#include <cmath>

typedef __attribute__((ext_vector_type(16))) _Float16 v16h;
typedef __attribute__((ext_vector_type(8)))  float    v8f;

#define B_    4
#define S_    2048
#define H_    8
#define D_    128
#define DM_   1024      // H_*D_
#define NROW_ (B_*S_)   // 8192
#define KT_   32        // keys per attention tile
#define NKT_  (S_/KT_)  // 64

#define WMMA(a,b,c) __builtin_amdgcn_wmma_f32_16x16x32_f16(false,(a),false,(b),(short)0,(c),false,false)

// ---- CDNA5 async global->LDS copy (ASYNCcnt path), per bridge doc via inline asm ----
__device__ __forceinline__ void async_load_b128(uint32_t lds_byte_off, const void* gaddr) {
  asm volatile("global_load_async_to_lds_b128 %0, %1, off"
               :: "v"(lds_byte_off), "v"(gaddr) : "memory");
}
__device__ __forceinline__ void wait_async0() {
  asm volatile("s_wait_asynccnt 0" ::: "memory");
}
// low 32 bits of a generic pointer to __shared__ == LDS byte offset on AMDGCN
__device__ __forceinline__ uint32_t ldsoff(const void* p) {
  return (uint32_t)(uintptr_t)p;
}

// A-operand (16x32 f16, MxK): lane m=l&15, h=l>>4; halves 0..7 -> k=8h+i, halves 8..15 -> k=16+8h+i
__device__ __forceinline__ v16h ld_frag_A(const _Float16* base, int stride) {
  int lane = threadIdx.x & 31;
  int m = lane & 15, h = lane >> 4;
  const _Float16* p = base + m * stride + 8 * h;
  union { uint4 u[2]; v16h v; } t;
  t.u[0] = *reinterpret_cast<const uint4*>(p);
  t.u[1] = *reinterpret_cast<const uint4*>(p + 16);
  return t.v;
}

// B-operand (32x16 f16, KxN) from an N-major (transposed) buffer: lane n=l&15, h=l>>4; halves 0..7 -> k=8h+i, halves 8..15 -> k=16+8h+i
__device__ __forceinline__ v16h ld_frag_B(const _Float16* base, int stride) {
  int lane = threadIdx.x & 31;
  int n = lane & 15, h = lane >> 4;
  const _Float16* p = base + n * stride + 8 * h;
  union { uint4 u[2]; v16h v; } t;
  t.u[0] = *reinterpret_cast<const uint4*>(p);
  t.u[1] = *reinterpret_cast<const uint4*>(p + 16);
  return t.v;
}

// dst[c*rows + r] = (f16) src[r*cols + c]
__global__ void transpose_cast_kernel(const float* __restrict__ src, _Float16* __restrict__ dst,
                                      int rows, int cols) {
  int idx = blockIdx.x * blockDim.x + threadIdx.x;
  if (idx >= rows * cols) return;
  int r = idx / cols, c = idx % cols;
  dst[(size_t)c * rows + r] = (_Float16)src[idx];
}

// C[row][o] = X[row][:128] . W[:, o] + bias[o], written f16 into per-head layout.
// vmode==0 : out[b][head][s][d]   (Q,K)
// vmode==1 : out[b][head][d][s]   (V transposed)
__global__ void __launch_bounds__(256)
proj_kernel(const float* __restrict__ X, const _Float16* __restrict__ WT,
            const float* __restrict__ bias, _Float16* __restrict__ out, int vmode) {
  __shared__ __align__(16) _Float16 Asub[64 * 136];
  __shared__ __align__(16) _Float16 Bsub[128 * 136];
  const int tid = threadIdx.x;
  const int nb  = blockIdx.x;   // head (128-col tile of W)
  const int mb  = blockIdx.y;   // 64-row tile

  // stage A: 64x128 f32 -> f16 (conversion: must go through VGPRs)
#pragma unroll
  for (int t = 0; t < 8; ++t) {
    int idx = tid + t * 256;            // 2048 float4
    int row = idx >> 5, c4 = idx & 31;
    float4 f = reinterpret_cast<const float4*>(X + (size_t)(mb * 64 + row) * 128)[c4];
    _Float16* d = Asub + row * 136 + c4 * 4;
    d[0] = (_Float16)f.x; d[1] = (_Float16)f.y; d[2] = (_Float16)f.z; d[3] = (_Float16)f.w;
  }
  // stage B^T slice async: rows o = nb*128 .. +128, each 128 halves
#pragma unroll
  for (int t = 0; t < 8; ++t) {
    int idx = tid + t * 256;            // 2048 uint4
    int row = idx >> 4, c = idx & 15;
    async_load_b128(ldsoff(Bsub) + (uint32_t)(row * 272 + c * 16),
                    WT + (size_t)(nb * 128 + row) * 128 + c * 8);
  }
  wait_async0();
  __syncthreads();

  const int w  = tid >> 5;
  const int wm = w >> 1;          // 0..3 : 16-row slice
  const int wn = w & 1;           // 0..1 : 64-col slice
  const int lane = tid & 31;
  const int nl = lane & 15, hh = lane >> 4;

  v16h afrag[4];
#pragma unroll
  for (int ks = 0; ks < 4; ++ks)
    afrag[ks] = ld_frag_A(Asub + wm * 16 * 136 + ks * 32, 136);

  v8f acc[4] = {};
#pragma unroll
  for (int nt = 0; nt < 4; ++nt) {
#pragma unroll
    for (int ks = 0; ks < 4; ++ks) {
      v16h bfrag = ld_frag_B(Bsub + (wn * 64 + nt * 16) * 136 + ks * 32, 136);
      acc[nt] = WMMA(afrag[ks], bfrag, acc[nt]);
    }
  }

#pragma unroll
  for (int nt = 0; nt < 4; ++nt) {
    int d = wn * 64 + nt * 16 + nl;
    float bb = bias[nb * 128 + d];
#pragma unroll
    for (int r = 0; r < 8; ++r) {
      int mrow = mb * 64 + wm * 16 + r + 8 * hh;
      int b = mrow >> 11, s = mrow & 2047;
      float val = acc[nt][r] + bb;
      size_t o;
      if (vmode) o = (((size_t)b * H_ + nb) * D_ + d) * S_ + s;   // [b][h][d][s]
      else       o = (((size_t)b * H_ + nb) * S_ + s) * D_ + d;   // [b][h][s][d]
      out[o] = (_Float16)val;
    }
  }
}

// Flash attention: 4 waves x 16 q-rows, 32-key tiles, async double-buffered K/V staging.
__global__ void __launch_bounds__(128)
attn_kernel(const _Float16* __restrict__ Qh, const _Float16* __restrict__ Kh,
            const _Float16* __restrict__ VhT, _Float16* __restrict__ AO) {
  __shared__ __align__(16) _Float16 Ksub[2][KT_ * 136];   // [key][d]
  __shared__ __align__(16) _Float16 Vsub[2][128 * 40];    // [d][key]  (V^T)
  __shared__ __align__(16) _Float16 Psub[64 * 40];        // [qrow][key]
  const int tid = threadIdx.x;
  const int qt  = blockIdx.x;     // 64-row query tile
  const int bh  = blockIdx.y;     // b*H + head
  const int w   = tid >> 5;
  const int lane = tid & 31;
  const int nl = lane & 15, hh = lane >> 4;
  const float scale = 0.08838834764831845f;  // 1/sqrt(128)

  const _Float16* qbase = Qh + ((size_t)bh * S_ + qt * 64 + w * 16) * D_;
  v16h qfrag[4];
#pragma unroll
  for (int ks = 0; ks < 4; ++ks) qfrag[ks] = ld_frag_A(qbase + ks * 32, D_);

  v8f o_acc[8] = {};
  float m_i[8], l_i[8];
#pragma unroll
  for (int r = 0; r < 8; ++r) { m_i[r] = -INFINITY; l_i[r] = 0.f; }

  const _Float16* kbase = Kh  + (size_t)bh * S_ * D_;
  const _Float16* vbase = VhT + (size_t)bh * D_ * S_;

  const uint32_t kldsbase[2] = { ldsoff(&Ksub[0][0]), ldsoff(&Ksub[1][0]) };
  const uint32_t vldsbase[2] = { ldsoff(&Vsub[0][0]), ldsoff(&Vsub[1][0]) };

  auto issue_tile = [&](int kt, int buf) {
#pragma unroll
    for (int t = 0; t < 4; ++t) {                 // K tile: 32 rows x 16 uint4
      int idx = tid + t * 128;
      int row = idx >> 4, c = idx & 15;
      async_load_b128(kldsbase[buf] + (uint32_t)(row * 272 + c * 16),
                      kbase + (size_t)(kt * KT_ + row) * D_ + c * 8);
    }
#pragma unroll
    for (int t = 0; t < 4; ++t) {                 // V^T tile: 128 rows x 4 uint4
      int idx = tid + t * 128;
      int row = idx >> 2, c = idx & 3;
      async_load_b128(vldsbase[buf] + (uint32_t)(row * 80 + c * 16),
                      vbase + (size_t)row * S_ + kt * KT_ + c * 8);
    }
  };

  issue_tile(0, 0);

  for (int kt = 0; kt < NKT_; ++kt) {
    const int cur = kt & 1;
    wait_async0();          // this wave's tile-kt async copies landed in LDS
    __syncthreads();        // ... and every other wave's too; tile kt-1 compute done
    if (kt + 1 < NKT_) issue_tile(kt + 1, cur ^ 1);
    if (kt + 6 < NKT_) {    // GL2 prefetch ~6 tiles ahead
      __builtin_prefetch(kbase + (size_t)(kt + 6) * KT_ * D_ + tid * 32, 0, 3);
      __builtin_prefetch(vbase + (size_t)tid * S_ + (kt + 6) * KT_, 0, 3);
    }

    // S = Q . K^T  (16 q-rows x 32 keys per wave)
    v8f sacc[2] = {};
#pragma unroll
    for (int ks = 0; ks < 4; ++ks) {
#pragma unroll
      for (int nt = 0; nt < 2; ++nt) {
        v16h bfrag = ld_frag_B(&Ksub[cur][nt * 16 * 136 + ks * 32], 136);
        sacc[nt] = WMMA(qfrag[ks], bfrag, sacc[nt]);
      }
    }
#pragma unroll
    for (int nt = 0; nt < 2; ++nt)
#pragma unroll
      for (int r = 0; r < 8; ++r) sacc[nt][r] *= scale;

    // online softmax; row m = r + 8*hh lives in one 16-lane half -> shfl_xor masks 1..8
#pragma unroll
    for (int r = 0; r < 8; ++r) {
      float t = fmaxf(sacc[0][r], sacc[1][r]);
#pragma unroll
      for (int off = 1; off < 16; off <<= 1) t = fmaxf(t, __shfl_xor(t, off, 32));
      float mnew  = fmaxf(m_i[r], t);
      float alpha = __expf(m_i[r] - mnew);
      m_i[r] = mnew;
      l_i[r] *= alpha;
#pragma unroll
      for (int ot = 0; ot < 8; ++ot) o_acc[ot][r] *= alpha;
      float ps = 0.f;
#pragma unroll
      for (int nt = 0; nt < 2; ++nt) {
        float p = __expf(sacc[nt][r] - mnew);
        ps += p;
        Psub[(w * 16 + r + 8 * hh) * 40 + nt * 16 + nl] = (_Float16)p;
      }
#pragma unroll
      for (int off = 1; off < 16; off <<= 1) ps += __shfl_xor(ps, off, 32);
      l_i[r] += ps;
    }
    // P rows are wave-private; DS ops in-order per wave -> compiler barrier only
    asm volatile("" ::: "memory");

    // O += P . V  (single K=32 step; B-operand from V^T tile)
    {
      v16h a2 = ld_frag_A(&Psub[w * 16 * 40], 40);
#pragma unroll
      for (int ot = 0; ot < 8; ++ot) {
        v16h bfrag = ld_frag_B(&Vsub[cur][ot * 16 * 40], 40);
        o_acc[ot] = WMMA(a2, bfrag, o_acc[ot]);
      }
    }
  }

  const int b = bh >> 3, head = bh & 7;
#pragma unroll
  for (int r = 0; r < 8; ++r) {
    float inv = 1.0f / l_i[r];
    int s = qt * 64 + w * 16 + r + 8 * hh;
    _Float16* dst = AO + ((size_t)b * S_ + s) * DM_ + head * D_;
#pragma unroll
    for (int ot = 0; ot < 8; ++ot)
      dst[ot * 16 + nl] = (_Float16)(o_acc[ot][r] * inv);
  }
}

// out[row][o] = AO[row][:1024] . Wo[:, o] + bo[o]  (fp32), async double-buffered K slabs of 32
__global__ void __launch_bounds__(256)
outproj_kernel(const _Float16* __restrict__ AO, const _Float16* __restrict__ WoT,
               const float* __restrict__ bo, float* __restrict__ out) {
  __shared__ __align__(16) _Float16 Asub[2][128 * 40];
  __shared__ __align__(16) _Float16 Bsub[2][128 * 40];
  const int tid = threadIdx.x;
  const int mb  = blockIdx.x;      // 128-row tile
  const int w   = tid >> 5;
  const int lane = tid & 31;
  const int nl = lane & 15, hh = lane >> 4;

  const uint32_t aldsbase[2] = { ldsoff(&Asub[0][0]), ldsoff(&Asub[1][0]) };
  const uint32_t bldsbase[2] = { ldsoff(&Bsub[0][0]), ldsoff(&Bsub[1][0]) };

  auto issue_slab = [&](int kt, int buf) {
#pragma unroll
    for (int t = 0; t < 2; ++t) {                 // 128 rows x 4 uint4 each
      int idx = tid + t * 256;
      int row = idx >> 2, c = idx & 3;
      async_load_b128(aldsbase[buf] + (uint32_t)(row * 80 + c * 16),
                      AO + (size_t)(mb * 128 + row) * DM_ + kt * 32 + c * 8);
      async_load_b128(bldsbase[buf] + (uint32_t)(row * 80 + c * 16),
                      WoT + (size_t)row * DM_ + kt * 32 + c * 8);
    }
  };

  issue_slab(0, 0);

  v8f acc[8] = {};
  for (int kt = 0; kt < 32; ++kt) {   // K = 1024 in 32-wide slabs
    const int cur = kt & 1;
    wait_async0();
    __syncthreads();
    if (kt + 1 < 32) issue_slab(kt + 1, cur ^ 1);
    if (kt + 6 < 32) {
      int r2 = tid & 127;
      if (tid < 128) __builtin_prefetch(AO  + (size_t)(mb * 128 + r2) * DM_ + (kt + 6) * 32, 0, 3);
      else           __builtin_prefetch(WoT + (size_t)r2 * DM_ + (kt + 6) * 32, 0, 3);
    }
    v16h af = ld_frag_A(&Asub[cur][w * 16 * 40], 40);
#pragma unroll
    for (int nt = 0; nt < 8; ++nt) {
      v16h bf = ld_frag_B(&Bsub[cur][nt * 16 * 40], 40);
      acc[nt] = WMMA(af, bf, acc[nt]);
    }
  }

#pragma unroll
  for (int nt = 0; nt < 8; ++nt) {
    int o = nt * 16 + nl;
    float bb = bo[o];
#pragma unroll
    for (int r = 0; r < 8; ++r) {
      int row = mb * 128 + w * 16 + r + 8 * hh;
      out[(size_t)row * D_ + o] = acc[nt][r] + bb;
    }
  }
}

extern "C" void kernel_launch(void* const* d_in, const int* in_sizes, int n_in,
                              void* d_out, int out_size, void* d_ws, size_t ws_size,
                              hipStream_t stream) {
  (void)in_sizes; (void)n_in; (void)out_size; (void)ws_size;
  const float* q  = (const float*)d_in[0];
  const float* k  = (const float*)d_in[1];
  const float* v  = (const float*)d_in[2];
  const float* Wq = (const float*)d_in[3];
  const float* bq = (const float*)d_in[4];
  const float* Wk = (const float*)d_in[5];
  const float* bk = (const float*)d_in[6];
  const float* Wv = (const float*)d_in[7];
  const float* bv = (const float*)d_in[8];
  const float* Wo = (const float*)d_in[9];
  const float* bo = (const float*)d_in[10];
  float* out = (float*)d_out;

  char* wsp = (char*)d_ws;
  size_t off = 0;
  auto alloc = [&](size_t bytes) -> char* {
    char* p = wsp + off;
    off += (bytes + 255) & ~(size_t)255;
    return p;
  };
  _Float16* WqT = (_Float16*)alloc((size_t)DM_ * D_ * 2);
  _Float16* WkT = (_Float16*)alloc((size_t)DM_ * D_ * 2);
  _Float16* WvT = (_Float16*)alloc((size_t)DM_ * D_ * 2);
  _Float16* WoT = (_Float16*)alloc((size_t)D_ * DM_ * 2);
  _Float16* Qh  = (_Float16*)alloc((size_t)B_ * H_ * S_ * D_ * 2);
  _Float16* Kh  = (_Float16*)alloc((size_t)B_ * H_ * S_ * D_ * 2);
  _Float16* VhT = (_Float16*)alloc((size_t)B_ * H_ * D_ * S_ * 2);
  _Float16* AO  = (_Float16*)alloc((size_t)B_ * S_ * DM_ * 2);

  const int NT = D_ * DM_;  // 131072
  transpose_cast_kernel<<<(NT + 255) / 256, 256, 0, stream>>>(Wq, WqT, D_, DM_);
  transpose_cast_kernel<<<(NT + 255) / 256, 256, 0, stream>>>(Wk, WkT, D_, DM_);
  transpose_cast_kernel<<<(NT + 255) / 256, 256, 0, stream>>>(Wv, WvT, D_, DM_);
  transpose_cast_kernel<<<(NT + 255) / 256, 256, 0, stream>>>(Wo, WoT, DM_, D_);

  dim3 pg(H_, NROW_ / 64);   // (8, 128)
  proj_kernel<<<pg, 256, 0, stream>>>(q, WqT, bq, Qh, 0);
  proj_kernel<<<pg, 256, 0, stream>>>(k, WkT, bk, Kh, 0);
  proj_kernel<<<pg, 256, 0, stream>>>(v, WvT, bv, VhT, 1);

  attn_kernel<<<dim3(S_ / 64, B_ * H_), 128, 0, stream>>>(Qh, Kh, VhT, AO);

  outproj_kernel<<<NROW_ / 128, 256, 0, stream>>>(AO, WoT, bo, out);
}
